// FeatureEnhance_86603720556862
// MI455X (gfx1250) — matrix operand order repaired and checked by reading, not compiled, not run on hardware
//
#include <hip/hip_runtime.h>

#define DEVI __device__ __forceinline__
#define GLOBAL_AS __attribute__((address_space(1)))
#define LDS_AS    __attribute__((address_space(3)))

typedef __attribute__((ext_vector_type(16))) __bf16 v16bf;
typedef __attribute__((ext_vector_type(8)))  float  v8f;
typedef int v4i_ __attribute__((vector_size(16)));
typedef unsigned short u16;

static constexpr int Bc = 2, C = 256, H = 64, W_ = 64, N = 4096, CQ = 32;

#if defined(__gfx1250__) && __has_builtin(__builtin_amdgcn_global_load_async_to_lds_b128)
#define HAVE_ASYNC_LDS 1
#endif

DEVI u16 f2bf(float f){
  __bf16 h = (__bf16)f;              // native f32->bf16 cvt
  u16 b; __builtin_memcpy(&b, &h, 2);
  return b;
}

union F16x16 { v16bf v; uint4 q[2]; u16 u[16]; };

// A-fragment (16x32 bf16, row-major source, K contiguous per row):
// lane(L): row M=L%16; elems 0..7 -> K = half*8+e ; elems 8..15 -> K = 16+half*8+(e-8)
DEVI v16bf load_a_frag(const u16* row, int half){
  F16x16 t;
  t.q[0] = *(const uint4*)(row + half*8);
  t.q[1] = *(const uint4*)(row + 16 + half*8);
  return t.v;
}
// B-fragment (32x16 bf16, column pointer with K contiguous):
// lane(L): col N=L%16; elems 0..7 -> K = half*8+e ; elems 8..15 -> K = 16+half*8+(e-8)
DEVI v16bf load_b_frag(const u16* col, int half){
  F16x16 t;
  t.q[0] = *(const uint4*)(col + half*8);
  t.q[1] = *(const uint4*)(col + 16 + half*8);
  return t.v;
}

DEVI v8f zero8(){ v8f z;
#pragma unroll
  for (int i=0;i<8;++i) z[i]=0.f; return z; }

DEVI v8f wmma_bf(v16bf a, v16bf b, v8f c){
  return __builtin_amdgcn_wmma_f32_16x16x32_bf16(false, a, false, b, (short)0, c, false, false);
}

// ---- async 16B global->LDS copy (with sync fallback) ----
DEVI void cp16_async(const u16* g, u16* l){
#if defined(HAVE_ASYNC_LDS)
  __builtin_amdgcn_global_load_async_to_lds_b128((GLOBAL_AS v4i_*)g,
                                                 (LDS_AS v4i_*)l, 0, 0);
#else
  *(uint4*)l = *(const uint4*)g;
#endif
}
DEVI void wait_async(){
#if defined(HAVE_ASYNC_LDS)
#if __has_builtin(__builtin_amdgcn_s_wait_asynccnt)
  __builtin_amdgcn_s_wait_asynccnt(0);
#else
  asm volatile("s_wait_asynccnt 0x0" ::: "memory");
#endif
#endif
}
// stage one 64-row K chunk (64 x 32 u16 = 4KB, contiguous) into LDS
DEVI void stage_k(const u16* src, u16* dst, int tid){
  const int offs = tid * 16;           // 128 threads x 32B
  cp16_async(src + offs,     dst + offs);
  cp16_async(src + offs + 8, dst + offs + 8);
}

// ---------------- elementwise bf16 copy of the 4 input streams ----------------
__global__ __launch_bounds__(256) void k_xcvt(const float* r, const float* g,
                                              const float* b, const float* a, u16* Xb){
  long i = ((long)blockIdx.x*256 + threadIdx.x)*4;
  const long strm = (long)Bc*C*N;
  int s = (int)(i / strm);
  long rem = i - (long)s*strm;
  const float* xs = (s==0)? r : (s==1)? g : (s==2)? b : a;
  float4 f = *(const float4*)(xs + rem);
  u16* o = Xb + i;
  o[0]=f2bf(f.x); o[1]=f2bf(f.y); o[2]=f2bf(f.z); o[3]=f2bf(f.w);
}

// ---------------- prepack conv weights: Wdb[khw][oc][ic] bf16 ----------------
__global__ __launch_bounds__(256) void k_wprep(const float* Wd, u16* Wdb){
  int i = blockIdx.x*256 + threadIdx.x;          // 9*256*256
  int khw = i / (C*C);
  int rem = i - khw*(C*C);
  int oc = rem / C, ic = rem - (rem/C)*C;
  Wdb[i] = f2bf(Wd[(oc*C + ic)*9 + khw]);
}

// ---------------- QKV projection GEMM: [320,256] x [256,N] per sb ----------------
__global__ __launch_bounds__(128) void k_qkv(
    const float* r, const float* g, const float* bl, const float* al,
    const float* Wq, const float* bq, const float* Wk, const float* bk,
    const float* Wv, const float* bv,
    u16* Qb, u16* Kb, u16* Vb)
{
  __shared__ u16 lx[64][32];                     // x tile transposed [n][c] bf16
  const int tid = threadIdx.x, wave = tid>>5, lane = tid&31, half = lane>>4, l16 = lane&15;
  const int n0 = blockIdx.x*64, octile = blockIdx.y, sb = blockIdx.z;
  const int s = sb >> 1, bb = sb & 1;
  const float* xp = ((s==0)? r : (s==1)? g : (s==2)? bl : al) + (long)bb*C*N;

  const int ocr = octile*16 + l16;               // A row handled by this lane
  const float* wr = (ocr<32) ? (Wq + ocr*C) : (ocr<64) ? (Wk + (ocr-32)*C) : (Wv + (ocr-64)*C);

  const int c_l = tid >> 2, nb_l = (tid & 3) * 16;
  v8f acc = zero8();
  for (int c0 = 0; c0 < C; c0 += 32){
    __syncthreads();
    const float* src = xp + (long)(c0 + c_l)*N + n0 + nb_l;
#pragma unroll
    for (int j = 0; j < 16; j += 4){
      float4 f = *(const float4*)(src + j);
      lx[nb_l+j+0][c_l] = f2bf(f.x);
      lx[nb_l+j+1][c_l] = f2bf(f.y);
      lx[nb_l+j+2][c_l] = f2bf(f.z);
      lx[nb_l+j+3][c_l] = f2bf(f.w);
    }
    __syncthreads();
    F16x16 aw;
#pragma unroll
    for (int e = 0; e < 8; ++e){
      aw.u[e]   = f2bf(wr[c0 + half*8 + e]);
      aw.u[8+e] = f2bf(wr[c0 + 16 + half*8 + e]);
    }
    v16bf bf_ = load_b_frag(&lx[wave*16 + l16][0], half);
    acc = wmma_bf(aw.v, bf_, acc);
  }
  const int n = n0 + wave*16 + l16;
#pragma unroll
  for (int v = 0; v < 8; ++v){
    int ocg = octile*16 + v + 8*half;
    float bias = (ocg<32)? bq[ocg] : (ocg<64)? bk[ocg-32] : bv[ocg-64];
    float val = acc[v] + bias;
    if (ocg < 32)      Qb[(long)sb*N*CQ + (long)n*CQ + ocg] = f2bf(val);         // [n][cq]
    else if (ocg < 64) Kb[(long)sb*N*CQ + (long)n*CQ + (ocg-32)] = f2bf(val);    // [n][cq]
    else               Vb[(long)sb*C*N + (long)(ocg-64)*N + n] = f2bf(val);      // [c][n]
  }
}

// ---------------- CAM energy: E = X * X^T  (256x256, k=4096) ----------------
__global__ __launch_bounds__(128) void k_came(const u16* Xb, float* ecam){
  const int tid = threadIdx.x, wave = tid>>5, lane = tid&31, half = lane>>4, l16 = lane&15;
  const int sb = blockIdx.z;
  const int tile = blockIdx.x*4 + wave;
  const int ct = tile >> 4, dt = tile & 15;
  const u16* X = Xb + (long)sb*C*N;
  const u16* rowA = X + (long)(ct*16 + l16)*N;
  const u16* rowB = X + (long)(dt*16 + l16)*N;
  v8f acc = zero8();
  for (int n0 = 0; n0 < N; n0 += 32){
    __builtin_prefetch(rowA + n0 + 512, 0, 1);
    acc = wmma_bf(load_a_frag(rowA + n0, half), load_b_frag(rowB + n0, half), acc);
  }
#pragma unroll
  for (int v = 0; v < 8; ++v)
    ecam[(long)sb*C*C + (long)(ct*16 + v + 8*half)*C + dt*16 + l16] = acc[v];
}

// ---------------- CAM softmax rows + sum over streams (gamma folded) ----------------
__global__ __launch_bounds__(256) void k_camsoft(const float* ecam, const float* gamma_cam,
                                                 u16* attsumb){
  __shared__ float sred[8];
  const int d = threadIdx.x, wave = d>>5;
  const int c = blockIdx.x, bb = blockIdx.y;
  float acc = 0.f;
  for (int s = 0; s < 4; ++s){
    float e = ecam[(long)(s*Bc + bb)*C*C + (long)c*C + d];
    float m = e;
#pragma unroll
    for (int k=16;k>=1;k>>=1) m = fmaxf(m, __shfl_xor(m, k, 32));
    if ((d&31)==0) sred[wave]=m;
    __syncthreads();
    float rmax = sred[0];
#pragma unroll
    for (int i=1;i<8;++i) rmax = fmaxf(rmax, sred[i]);
    __syncthreads();
    float val = rmax - e;                 // energy_new
    float m2 = val;
#pragma unroll
    for (int k=16;k>=1;k>>=1) m2 = fmaxf(m2, __shfl_xor(m2, k, 32));
    if ((d&31)==0) sred[wave]=m2;
    __syncthreads();
    float vmax = sred[0];
#pragma unroll
    for (int i=1;i<8;++i) vmax = fmaxf(vmax, sred[i]);
    __syncthreads();
    float p = __expf(val - vmax);
    float sum = p;
#pragma unroll
    for (int k=16;k>=1;k>>=1) sum += __shfl_xor(sum, k, 32);
    if ((d&31)==0) sred[wave]=sum;
    __syncthreads();
    float tsum = 0.f;
#pragma unroll
    for (int i=0;i<8;++i) tsum += sred[i];
    __syncthreads();
    acc += p / tsum;
  }
  attsumb[(long)bb*C*C + (long)c*C + d] = f2bf(gamma_cam[0] * acc);
}

// ------- PAM pass 1: row max / sum-of-exp; K staged via async double buffer -------
__global__ __launch_bounds__(128) void k_stats(const u16* Qb, const u16* Kb,
                                               float* rowM, float* rowL){
  __shared__ u16 Ks[2][64][32];   // double-buffered K chunk (4KB each)
  const int tid = threadIdx.x, wave = tid>>5, lane = tid&31, half = lane>>4, l16 = lane&15;
  const int sb = blockIdx.z;
  const int nt = blockIdx.x*64 + wave*16;
  const u16* Q  = Qb + (long)sb*N*CQ;
  const u16* Kg = Kb + (long)sb*N*CQ;
  v16bf qf = load_a_frag(Q + (long)(nt + l16)*CQ, half);   // k = CQ = 32: one WMMA depth
  float rm[8], rl[8];
#pragma unroll
  for (int v=0;v<8;++v){ rm[v] = -3.0e38f; rl[v] = 0.f; }

  stage_k(Kg, &Ks[0][0][0], tid);                 // preload chunk 0
  for (int ch = 0; ch < N/64; ++ch){
    wait_async();
    __syncthreads();
    if (ch + 1 < N/64) stage_k(Kg + (ch+1)*64*CQ, &Ks[(ch+1)&1][0][0], tid);
    const u16* Kt = &Ks[ch&1][0][0];
#pragma unroll
    for (int mm = 0; mm < 4; ++mm){
      v16bf kf = load_b_frag(Kt + (mm*16 + l16)*CQ, half);
      v8f e = wmma_bf(qf, kf, zero8());
#pragma unroll
      for (int v=0;v<8;++v){
        float x = e[v];
        float mx = x;
#pragma unroll
        for (int k=8;k>=1;k>>=1) mx = fmaxf(mx, __shfl_xor(mx, k, 32)); // within 16-lane half
        float nm = fmaxf(rm[v], mx);
        float p = __expf(x - nm);
        float ps = p;
#pragma unroll
        for (int k=8;k>=1;k>>=1) ps += __shfl_xor(ps, k, 32);
        rl[v] = rl[v]*__expf(rm[v]-nm) + ps;
        rm[v] = nm;
      }
    }
  }
#pragma unroll
  for (int v=0;v<8;++v){
    if (l16 == v){
      rowM[(long)sb*N + nt + 8*half + v] = rm[v];
      rowL[(long)sb*N + nt + 8*half + v] = rl[v];
    }
  }
}

// ------- PAM pass 2 + CAM apply + combine -> Yb bf16 (K staged via async) -------
__global__ __launch_bounds__(128) void k_apply(
    const float* r, const float* g, const float* bl, const float* al,
    const u16* Qb, const u16* Kb, const u16* Vb, const u16* Xb,
    const float* rowM, const float* rowL, const u16* attsumb,
    const float* gamma_pam, u16* Yb)
{
  __shared__ u16 Ks[2][64][32];   // double-buffered K chunk
  __shared__ u16 P[4][16][32];    // per-wave probability tile [n][m] bf16
  __shared__ u16 XT[16][32];      // staged X^T block [n][d]
  const int tid = threadIdx.x, wave = tid>>5, lane = tid&31, half = lane>>4, l16 = lane&15;
  const int sb = blockIdx.z, nb = blockIdx.x*16;
  const int s = sb>>1, bb = sb&1;
  const u16* Q  = Qb + (long)sb*N*CQ;
  const u16* Kg = Kb + (long)sb*N*CQ;
  const u16* V  = Vb + (long)sb*C*N;
  v16bf qf = load_a_frag(Q + (long)(nb + l16)*CQ, half);
  float rm[8], rli[8];
#pragma unroll
  for (int v=0;v<8;++v){
    rm[v]  = rowM[(long)sb*N + nb + v + 8*half];
    rli[v] = 1.0f / rowL[(long)sb*N + nb + v + 8*half];
  }
  v8f pam[4], cam[4];
#pragma unroll
  for (int j=0;j<4;++j){ pam[j]=zero8(); cam[j]=zero8(); }

  stage_k(Kg, &Ks[0][0][0], tid);                 // preload chunk 0
  for (int ch = 0; ch < N/64; ++ch){
    wait_async();
    __syncthreads();
    if (ch + 1 < N/64) stage_k(Kg + (ch+1)*64*CQ, &Ks[(ch+1)&1][0][0], tid);
    const u16* Kt = &Ks[ch&1][0][0];
#pragma unroll
    for (int sub = 0; sub < 2; ++sub){
      const int m0 = ch*64 + sub*32;
      v16bf k0 = load_b_frag(Kt + (sub*32      + l16)*CQ, half);
      v16bf k1 = load_b_frag(Kt + (sub*32 + 16 + l16)*CQ, half);
      v8f e0 = wmma_bf(qf, k0, zero8());
      v8f e1 = wmma_bf(qf, k1, zero8());
#pragma unroll
      for (int v=0;v<8;++v){
        int nrow = v + 8*half;
        P[wave][nrow][l16]      = f2bf(__expf(e0[v]-rm[v])*rli[v]);
        P[wave][nrow][16 + l16] = f2bf(__expf(e1[v]-rm[v])*rli[v]);
      }
      // per-wave LDS transpose: LDS is in-order within a wave
      v16bf pf = load_b_frag(&P[wave][l16][0], half);
#pragma unroll
      for (int j=0;j<4;++j){
        const u16* vrow = V + (long)((wave*4+j)*16 + l16)*N + m0;
        pam[j] = wmma_bf(load_a_frag(vrow, half), pf, pam[j]);
      }
    }
  }

  for (int d0=0; d0<C; d0+=32){
    __syncthreads();
    {
      int d_l = tid & 31, ng = tid >> 5;
      const u16* src = Xb + (long)sb*C*N + (long)(d0 + d_l)*N + nb + ng*4;
#pragma unroll
      for (int jj=0;jj<4;++jj) XT[ng*4+jj][d_l] = src[jj];
    }
    __syncthreads();
    v16bf bfc = load_b_frag(&XT[l16][0], half);
#pragma unroll
    for (int j=0;j<4;++j){
      const u16* arow = attsumb + (long)bb*C*C + (long)((wave*4+j)*16 + l16)*C + d0;
      cam[j] = wmma_bf(load_a_frag(arow, half), bfc, cam[j]);
    }
  }

  const float gp = gamma_pam[0];
  const float* xp = ((s==0)? r : (s==1)? g : (s==2)? bl : al) + (long)bb*C*N;
#pragma unroll
  for (int j=0;j<4;++j){
#pragma unroll
    for (int v=0;v<8;++v){
      int c = (wave*4+j)*16 + v + 8*half;
      int n = nb + l16;
      float y = gp*pam[j][v] + cam[j][v] + 5.0f*xp[(long)c*N + n];
      Yb[(long)sb*C*N + (long)c*N + n] = f2bf(y);
    }
  }
}

// ---------------- 3x3 conv as implicit GEMM (k = 9 x 256) ----------------
__global__ __launch_bounds__(128) void k_conv(const u16* Yb, const u16* Wdb,
                                              const float* bd, float* out){
  __shared__ u16 Ys[64][32];      // shifted Y block [pixel][ic] bf16 (zero-padded)
  const int tid = threadIdx.x, wave = tid>>5, lane = tid&31, half = lane>>4, l16 = lane&15;
  const int yrow = blockIdx.x, octile = blockIdx.y, sb = blockIdx.z;
  const u16* Yp = Yb + (long)sb*C*N;
  v8f acc = zero8();
  const int ic_l = tid >> 2, pxb = (tid & 3) * 16;
  for (int khw=0; khw<9; ++khw){
    const int kh = khw/3, kw = khw - kh*3;
    const int yy = yrow + kh - 1;
    for (int ic0=0; ic0<C; ic0+=32){
      __syncthreads();
#pragma unroll
      for (int j=0;j<16;++j){
        int px = pxb + j;
        int xx = px + kw - 1;
        u16 val = 0;
        if (yy>=0 && yy<H && xx>=0 && xx<W_)
          val = Yp[(long)(ic0+ic_l)*N + yy*W_ + xx];
        Ys[px][ic_l] = val;
      }
      __syncthreads();
      v16bf bf_ = load_b_frag(&Ys[wave*16 + l16][0], half);
      const u16* arow = Wdb + (long)khw*C*C + (long)(octile*16 + l16)*C + ic0;
      acc = wmma_bf(load_a_frag(arow, half), bf_, acc);
    }
  }
#pragma unroll
  for (int v=0;v<8;++v){
    int oc = octile*16 + v + 8*half;
    out[(long)sb*C*N + (long)oc*N + yrow*W_ + wave*16 + l16] = acc[v] + bd[oc];
  }
}

extern "C" void kernel_launch(void* const* d_in, const int* in_sizes, int n_in,
                              void* d_out, int out_size, void* d_ws, size_t ws_size,
                              hipStream_t stream){
  (void)in_sizes; (void)n_in; (void)out_size; (void)ws_size;
  const float* r  = (const float*)d_in[0];
  const float* g  = (const float*)d_in[1];
  const float* bl = (const float*)d_in[2];
  const float* al = (const float*)d_in[3];
  const float* Wq = (const float*)d_in[4];
  const float* bq = (const float*)d_in[5];
  const float* Wk = (const float*)d_in[6];
  const float* bk = (const float*)d_in[7];
  const float* Wv = (const float*)d_in[8];
  const float* bv = (const float*)d_in[9];
  const float* gp = (const float*)d_in[10];
  const float* gc = (const float*)d_in[11];
  const float* Wd = (const float*)d_in[12];
  const float* bd = (const float*)d_in[13];

  char* w = (char*)d_ws;
  u16*   Qb   = (u16*)(w);                          //  2 MB  [sb][n][32]
  u16*   Kb   = (u16*)(w + (2ul<<20));              //  2 MB  [sb][n][32]
  u16*   Vb   = (u16*)(w + (4ul<<20));              // 16 MB  [sb][c][n]
  u16*   Xb   = (u16*)(w + (20ul<<20));             // 16 MB  [sb][c][n]
  float* ecam = (float*)(w + (36ul<<20));           //  2 MB  [sb][c][d]
  u16*   atts = (u16*)(w + (38ul<<20));             // 256 KB [b][c][d]
  float* rowM = (float*)(w + (39ul<<20));           // 128 KB
  float* rowL = (float*)(w + (39ul<<20) + (1ul<<18));
  u16*   Yb   = (u16*)(w + (40ul<<20));             // 16 MB  [sb][c][n]
  u16*   Wdb  = (u16*)(w + (56ul<<20));             // 1.2 MB [khw][oc][ic]
  float* out  = (float*)d_out;

  k_xcvt  <<<8192, 256, 0, stream>>>(r,g,bl,al,Xb);
  k_wprep <<<2304, 256, 0, stream>>>(Wd,Wdb);
  k_qkv   <<<dim3(64,20,8), 128, 0, stream>>>(r,g,bl,al,Wq,bq,Wk,bk,Wv,bv,Qb,Kb,Vb);
  k_came  <<<dim3(64,1,8),  128, 0, stream>>>(Xb,ecam);
  k_camsoft<<<dim3(256,2),  256, 0, stream>>>(ecam,gc,atts);
  k_stats <<<dim3(64,1,8),  128, 0, stream>>>(Qb,Kb,rowM,rowL);
  k_apply <<<dim3(256,1,8), 128, 0, stream>>>(r,g,bl,al,Qb,Kb,Vb,Xb,rowM,rowL,atts,gp,Yb);
  k_conv  <<<dim3(64,16,8), 128, 0, stream>>>(Yb,Wdb,bd,out);
}
